// EfficientCrossAttention_47493748359984
// MI455X (gfx1250) — matrix operand order repaired and checked by reading, not compiled, not run on hardware
//
#include <hip/hip_runtime.h>
#include <hip/hip_bf16.h>

// ---------------------------------------------------------------------------
// EfficientCrossAttention for MI455X (gfx1250, wave32, WMMA)
//   Q = X1*Wq + bq ; K = X2*Wk + bk ; V = X2*Wv + bv   (bf16 in workspace)
//   O = softmax(clip(Q K^T / sqrt(D))) V                (flash-style, f32 out)
// ---------------------------------------------------------------------------

typedef __bf16 bf16_t;
typedef __attribute__((ext_vector_type(16))) __bf16 bf16x16;
typedef __attribute__((ext_vector_type(8)))  __bf16 bf16x8;
typedef __attribute__((ext_vector_type(8)))  float  f32x8;
typedef __attribute__((ext_vector_type(4)))  float  f32x4;
typedef __attribute__((ext_vector_type(4)))  int    i32x4;

#if defined(__has_builtin)
#  if __has_builtin(__builtin_amdgcn_global_load_async_to_lds_b128)
#    define HAVE_ASYNC_LDS 1
#  endif
#  if __has_builtin(__builtin_amdgcn_s_wait_asynccnt)
#    define HAVE_WAIT_ASYNC 1
#  endif
#endif

typedef __attribute__((address_space(1))) i32x4* as1_i32x4p;
typedef __attribute__((address_space(3))) i32x4* as3_i32x4p;

union FragU { bf16x16 f; bf16x8 h[2]; };

// A-matrix 16x32 bf16 fragment (ISA 7.12.2): lane row = l%16,
// elements e -> K = 16*(e/8) + 8*(l/16) + e%8  => two contiguous 16B chunks.
__device__ __forceinline__ bf16x16 load_frag_a(const bf16_t* base, int stride,
                                               int row0, int k0, int lane) {
  const bf16_t* p = base + (size_t)(row0 + (lane & 15)) * stride + k0 + 8 * (lane >> 4);
  FragU u;
  u.h[0] = *(const bf16x8*)(p);
  u.h[1] = *(const bf16x8*)(p + 16);
  return u.f;
}

// B-matrix 32x16 bf16 fragment: lane col = l%16, elements e -> K = 16*(e/8) + 8*(l/16) + e%8
// => two contiguous 16B chunks. "row0" indexes the N dimension of a [N][K]-stored
// matrix (i.e. already transposed so K is contiguous).
__device__ __forceinline__ bf16x16 load_frag_b(const bf16_t* base, int stride,
                                               int row0, int k0, int lane) {
  const bf16_t* p = base + (size_t)(row0 + (lane & 15)) * stride + k0 + 8 * (lane >> 4);
  FragU u;
  u.h[0] = *(const bf16x8*)(p);
  u.h[1] = *(const bf16x8*)(p + 16);
  return u.f;
}

__device__ __forceinline__ f32x8 wmma_bf16(bf16x16 a, bf16x16 b, f32x8 c) {
  return __builtin_amdgcn_wmma_f32_16x16x32_bf16(false, a, false, b,
                                                 (short)0, c, false, false);
}

// ---------------------------------------------------------------------------
// Projection GEMM:  out[16384 x 1024](bf16) = X[16384 x 1024](f32) * W + bias
// Block tile 128x128, 8 waves of 32x64, K-slab 64 (16 WMMAs per barrier pair),
// fp32->bf16 conversion through padded LDS.
// transposed==1 stores V^T layout: out[(b*1024 + col)*2048 + row%2048].
// ---------------------------------------------------------------------------
__global__ __launch_bounds__(256)
void proj_gemm_kernel(const float* __restrict__ X, const float* __restrict__ W,
                      const float* __restrict__ bias, bf16_t* __restrict__ out,
                      int transposed) {
  __shared__ bf16_t Xs[128 * 72];   // [128 rows][64 K + 8 pad]
  __shared__ bf16_t Wt[128 * 72];   // [128 cols][64 K + 8 pad]  (W transposed)

  const int tid  = threadIdx.x;
  const int lane = tid & 31;
  const int w    = tid >> 5;
  const int row0 = blockIdx.y * 128;
  const int col0 = blockIdx.x * 128;
  const int wm   = (w & 3) * 32;   // wave row offset inside block tile
  const int wn   = (w >> 2) * 64;  // wave col offset inside block tile

  f32x8 acc[2][4];
  f32x8 zero = {};
#pragma unroll
  for (int i = 0; i < 2; ++i)
#pragma unroll
    for (int j = 0; j < 4; ++j) acc[i][j] = zero;

  for (int kk = 0; kk < 1024; kk += 64) {
    // ---- stage X tile 128x64 (fp32 -> bf16) ----
    {
      const int r = tid >> 1;
      const int c = (tid & 1) * 32;
      const float* src = X + (size_t)(row0 + r) * 1024 + kk + c;
      bf16_t* dst = &Xs[r * 72 + c];
#pragma unroll
      for (int g = 0; g < 2; ++g) {
        f32x4 v0 = *(const f32x4*)(src + g * 16 + 0);
        f32x4 v1 = *(const f32x4*)(src + g * 16 + 4);
        f32x4 v2 = *(const f32x4*)(src + g * 16 + 8);
        f32x4 v3 = *(const f32x4*)(src + g * 16 + 12);
        bf16x8 b0, b1;
#pragma unroll
        for (int i = 0; i < 4; ++i) {
          b0[i]     = (bf16_t)v0[i];
          b0[i + 4] = (bf16_t)v1[i];
          b1[i]     = (bf16_t)v2[i];
          b1[i + 4] = (bf16_t)v3[i];
        }
        *(bf16x8*)(dst + g * 16)     = b0;
        *(bf16x8*)(dst + g * 16 + 8) = b1;
      }
    }
    // ---- stage W tile 64x128 transposed -> Wt[128][64] (fp32 -> bf16) ----
    {
      const int k  = tid >> 2;            // 0..63
      const int n0 = (tid & 3) * 32;      // 0,32,64,96
      const float* src = W + (size_t)(kk + k) * 1024 + col0 + n0;
#pragma unroll
      for (int j = 0; j < 8; ++j) {
        f32x4 v = *(const f32x4*)(src + j * 4);
#pragma unroll
        for (int i = 0; i < 4; ++i) Wt[(n0 + j * 4 + i) * 72 + k] = (bf16_t)v[i];
      }
    }
    __syncthreads();

    // ---- 16 WMMAs per wave on the staged K=64 slab ----
#pragma unroll
    for (int kl = 0; kl < 64; kl += 32) {
      bf16x16 bfr[4];
#pragma unroll
      for (int nt = 0; nt < 4; ++nt)
        bfr[nt] = load_frag_b(Wt, 72, wn + nt * 16, kl, lane);
#pragma unroll
      for (int mt = 0; mt < 2; ++mt) {
        bf16x16 a = load_frag_a(Xs, 72, wm + mt * 16, kl, lane);
#pragma unroll
        for (int nt = 0; nt < 4; ++nt)
          acc[mt][nt] = wmma_bf16(a, bfr[nt], acc[mt][nt]);
      }
    }
    __syncthreads();
  }

  // ---- epilogue: bias add, bf16 store (optionally transposed for V) ----
#pragma unroll
  for (int mt = 0; mt < 2; ++mt) {
#pragma unroll
    for (int nt = 0; nt < 4; ++nt) {
      const int col = col0 + wn + nt * 16 + (lane & 15);
      const float bv = bias[col];
      const int rbase = row0 + wm + mt * 16 + 8 * (lane >> 4);
#pragma unroll
      for (int v = 0; v < 8; ++v) {
        const int r = rbase + v;
        const float val = acc[mt][nt][v] + bv;
        if (!transposed) {
          out[(size_t)r * 1024 + col] = (bf16_t)val;
        } else {
          const int b  = r >> 11;       // r / 2048
          const int rr = r & 2047;
          out[((size_t)(b * 1024 + col)) * 2048 + rr] = (bf16_t)val;
        }
      }
    }
  }
}

// ---------------------------------------------------------------------------
// Flash attention: block = 64 query rows of one batch, 16 waves (512 thr).
//   S phase : each wave computes one 16x16 tile of S(64x64) over D=1024.
//   Softmax : fully parallel; each wave owns 4 rows, 8 lanes/row, shfl_xor
//             reductions (wave32).
//   PV phase: each wave owns a 64-wide D slice of O (128 f32 accum VGPRs),
//             B-fragments stream from V^T (contiguous in key dim).
// ---------------------------------------------------------------------------
__global__ __launch_bounds__(512)
void flash_attn_kernel(const bf16_t* __restrict__ Q, const bf16_t* __restrict__ K,
                       const bf16_t* __restrict__ VT, float* __restrict__ out) {
  __shared__ bf16_t Qs[64 * 1032];  // 132KB, padded: 1032*2B = 129*16B
  __shared__ float  Ss[64 * 64];    // 16KB
  __shared__ bf16_t Ps[64 * 72];    // 9KB, padded: 72*2B = 9*16B
  __shared__ float  rowM[64], rowL[64], rowF[64];

  const int tid  = threadIdx.x;
  const int lane = tid & 31;
  const int w    = tid >> 5;
  const int b    = blockIdx.y;
  const int q0   = blockIdx.x * 64;
  const float scale = 0.03125f;  // 1/sqrt(1024)

  const bf16_t* Qb = Q  + ((size_t)b * 2048 + q0) * 1024;
  const bf16_t* Kb = K  + (size_t)b * 2048 * 1024;
  const bf16_t* Vb = VT + (size_t)b * 1024 * 2048;

  // ---- stage Q tile 64x1024 into LDS (async DMA if available) ----
#if defined(HAVE_ASYNC_LDS)
  for (int i = tid; i < 8192; i += 512) {   // 8192 chunks of 16B
    const int r = i >> 7;
    const int c = (i & 127) * 8;
    __builtin_amdgcn_global_load_async_to_lds_b128(
        (as1_i32x4p)(Qb + (size_t)r * 1024 + c),
        (as3_i32x4p)(&Qs[r * 1032 + c]), 0, 0);
  }
#  if defined(HAVE_WAIT_ASYNC)
  __builtin_amdgcn_s_wait_asynccnt(0);
#  else
  asm volatile("s_wait_asynccnt 0x0" ::: "memory");
#  endif
#else
  for (int i = tid; i < 8192; i += 512) {
    const int r = i >> 7;
    const int c = (i & 127) * 8;
    *(bf16x8*)&Qs[r * 1032 + c] = *(const bf16x8*)(Qb + (size_t)r * 1024 + c);
  }
#endif
  if (tid < 64) { rowM[tid] = -1e30f; rowL[tid] = 0.f; }

  f32x8 o[4][4];
  f32x8 zero = {};
#pragma unroll
  for (int i = 0; i < 4; ++i)
#pragma unroll
    for (int j = 0; j < 4; ++j) o[i][j] = zero;

  __syncthreads();

  const int smt = w & 3;   // this wave's S tile (row, col)
  const int snt = w >> 2;

  for (int kb = 0; kb < 2048; kb += 64) {
    // ---- prefetch next K / V^T key block into L2/WGP$ ----
    if (kb + 64 < 2048) {
      const bf16_t* kp = Kb + (size_t)(kb + 64 + (tid >> 3)) * 1024 + (tid & 7) * 128;
      __builtin_prefetch((const void*)kp, 0, 1);
      const bf16_t* vp = Vb + (size_t)(tid * 2) * 2048 + kb + 64;
      __builtin_prefetch((const void*)vp, 0, 1);
    }

    // ---- S = Q K^T : 32 k-steps over D, one 16x16 tile per wave ----
    f32x8 s = zero;
#pragma unroll 4
    for (int kk = 0; kk < 1024; kk += 32) {
      bf16x16 a  = load_frag_a(Qs, 1032, smt * 16, kk, lane);
      bf16x16 bb = load_frag_b(Kb, 1024, kb + snt * 16, kk, lane);
      s = wmma_bf16(a, bb, s);
    }
    {
      const int col   = snt * 16 + (lane & 15);
      const int rbase = smt * 16 + 8 * (lane >> 4);
#pragma unroll
      for (int v = 0; v < 8; ++v) Ss[(rbase + v) * 64 + col] = s[v];
    }
    __syncthreads();

    // ---- online softmax: wave w owns rows [4w,4w+4), 8 lanes per row ----
    {
      const int r  = w * 4 + (lane >> 3);
      const int c0 = (lane & 7) * 8;
      const float m_old = rowM[r];
      float mx = m_old;
      float xs[8];
#pragma unroll
      for (int i = 0; i < 8; ++i) {
        float x = Ss[r * 64 + c0 + i] * scale;
        x = fminf(fmaxf(x, -100.f), 100.f);
        xs[i] = x;
        mx = fmaxf(mx, x);
      }
      mx = fmaxf(mx, __shfl_xor(mx, 1));
      mx = fmaxf(mx, __shfl_xor(mx, 2));
      mx = fmaxf(mx, __shfl_xor(mx, 4));
      float sum = 0.f;
      bf16x8 pv;
#pragma unroll
      for (int i = 0; i < 8; ++i) {
        float p = __expf(xs[i] - mx);
        sum += p;
        pv[i] = (bf16_t)p;
      }
      *(bf16x8*)&Ps[r * 72 + c0] = pv;
      sum += __shfl_xor(sum, 1);
      sum += __shfl_xor(sum, 2);
      sum += __shfl_xor(sum, 4);
      if ((lane & 7) == 0) {
        const float f = __expf(m_old - mx);
        rowM[r] = mx;
        rowF[r] = f;
        rowL[r] = rowL[r] * f + sum;
      }
    }
    __syncthreads();

    // ---- rescale O, then O += P * V  (K=64 -> two k-steps) ----
#pragma unroll
    for (int mt = 0; mt < 4; ++mt) {
      const int rbase = mt * 16 + 8 * (lane >> 4);
#pragma unroll
      for (int v = 0; v < 8; ++v) {
        const float f = rowF[rbase + v];
#pragma unroll
        for (int nt = 0; nt < 4; ++nt) o[mt][nt][v] *= f;
      }
    }
#pragma unroll
    for (int kk2 = 0; kk2 < 64; kk2 += 32) {
      bf16x16 afr[4];
#pragma unroll
      for (int mt = 0; mt < 4; ++mt)
        afr[mt] = load_frag_a(Ps, 72, mt * 16, kk2, lane);
#pragma unroll
      for (int nt = 0; nt < 4; ++nt) {
        bf16x16 bb = load_frag_b(Vb, 2048, w * 64 + nt * 16, kb + kk2, lane);
#pragma unroll
        for (int mt = 0; mt < 4; ++mt)
          o[mt][nt] = wmma_bf16(afr[mt], bb, o[mt][nt]);
      }
    }
    __syncthreads();
  }

  // ---- epilogue: O / rowsum, f32 store; wave w owns D slice [64w, 64w+64) ----
#pragma unroll
  for (int mt = 0; mt < 4; ++mt) {
    const int rbase = mt * 16 + 8 * (lane >> 4);
#pragma unroll
    for (int v = 0; v < 8; ++v) {
      const int r = rbase + v;
      const float linv = 1.0f / rowL[r];
      float* op = out + ((size_t)b * 2048 + q0 + r) * 1024 + w * 64 + (lane & 15);
#pragma unroll
      for (int nt = 0; nt < 4; ++nt) op[nt * 16] = o[mt][nt][v] * linv;
    }
  }
}

// ---------------------------------------------------------------------------
extern "C" void kernel_launch(void* const* d_in, const int* in_sizes, int n_in,
                              void* d_out, int out_size, void* d_ws, size_t ws_size,
                              hipStream_t stream) {
  (void)in_sizes; (void)n_in; (void)out_size; (void)ws_size;
  const float* input_1 = (const float*)d_in[0];
  const float* input_2 = (const float*)d_in[1];
  const float* Wq = (const float*)d_in[2];
  const float* bq = (const float*)d_in[3];
  const float* Wk = (const float*)d_in[4];
  const float* bk = (const float*)d_in[5];
  const float* Wv = (const float*)d_in[6];
  const float* bv = (const float*)d_in[7];

  bf16_t* Qb = (bf16_t*)d_ws;                       // [8*2048][1024] bf16
  bf16_t* Kb = Qb + (size_t)16384 * 1024;           // [8*2048][1024] bf16
  bf16_t* Vt = Kb + (size_t)16384 * 1024;           // [8][1024][2048] bf16 (V^T)

  dim3 gg(8, 128), blk(256);
  proj_gemm_kernel<<<gg, blk, 0, stream>>>(input_1, Wq, bq, Qb, 0);
  proj_gemm_kernel<<<gg, blk, 0, stream>>>(input_2, Wk, bk, Kb, 0);
  proj_gemm_kernel<<<gg, blk, 0, stream>>>(input_2, Wv, bv, Vt, 1);

  flash_attn_kernel<<<dim3(32, 8), 512, 0, stream>>>(Qb, Kb, Vt, (float*)d_out);
}
